// _PointNet2SAModuleBase_89550068122184
// MI455X (gfx1250) — matrix operand order repaired and checked by reading, compiled, not run on hardware
//
#include <hip/hip_runtime.h>
#include <math.h>

typedef __attribute__((ext_vector_type(16))) _Float16 v16h;
typedef __attribute__((ext_vector_type(8)))  _Float16 v8h;
typedef __attribute__((ext_vector_type(8)))  float    v8f;

// Problem constants (match reference)
constexpr int Bc = 2, Nc = 4096, Mc = 1024, K1c = 32, K2c = 16;

// ---------------------------------------------------------------------------
// Generic WMMA GEMM:  C[rows x Cout] = ReLU(A[rows x Kpad] * W[Cout x Kpad]^T + bias)
// rows % 16 == 0, Kpad % 32 == 0, Cout % 16 == 0. A, W fp16; bias, C fp32.
// One wave per 16x16 tile; EXEC is all-ones (guard is wave-uniform).
// ---------------------------------------------------------------------------
__global__ void gemm_relu_wmma(const _Float16* __restrict__ A,
                               const _Float16* __restrict__ W,
                               const float* __restrict__ bias,
                               float* __restrict__ C,
                               int rows, int Kpad, int Cout) {
  int rowTiles = rows >> 4, colTiles = Cout >> 4;
  int wave = blockIdx.x * (blockDim.x >> 5) + (threadIdx.x >> 5);
  if (wave >= rowTiles * colTiles) return;          // wave-uniform exit
  int rt = wave / colTiles, ct = wave % colTiles;
  int lane = threadIdx.x & 31;
  int l = lane & 15, hi = lane >> 4;

  const _Float16* arow = A + (size_t)(rt * 16 + l) * Kpad;
  const _Float16* wrow = W + (size_t)(ct * 16 + l) * Kpad + hi * 16;

  v8f acc = {};
  for (int kk = 0; kk < Kpad; kk += 32) {
    // A fragment: lane row l holds K{kk + hi*16 .. +15} (contiguous)
    v16h af = *(const v16h*)(arow + kk + hi * 16);
    // B fragment: lane n holds W[colTile+n][kk + hi*16 .. +15] (contiguous)
    v16h bf = *(const v16h*)(wrow + kk);
    acc = __builtin_amdgcn_wmma_f32_16x16x32_f16(false, af, false, bf,
                                                 (short)0, acc, false, false);
  }
  int col = ct * 16 + l;
  float bv = bias[col];
  int rbase = rt * 16 + hi * 8;
#pragma unroll
  for (int j = 0; j < 8; ++j) {
    float v = acc[j] + bv;
    C[(size_t)(rbase + j) * Cout + col] = v > 0.f ? v : 0.f;
  }
}

// ---------------------------------------------------------------------------
// Weight packing: (Cout x Kin) f32  ->  (Cout x Kpad) f16, zero padded
// ---------------------------------------------------------------------------
__global__ void pack_w(const float* __restrict__ src, _Float16* __restrict__ dst,
                       int Cout, int Kin, int Kpad) {
  int i = blockIdx.x * blockDim.x + threadIdx.x;
  if (i >= Cout * Kpad) return;
  int co = i / Kpad, ki = i % Kpad;
  dst[i] = (ki < Kin) ? (_Float16)src[co * Kin + ki] : (_Float16)0.f;
}

__global__ void f32_to_f16(const float* __restrict__ src, _Float16* __restrict__ dst, int n) {
  int i = blockIdx.x * blockDim.x + threadIdx.x;
  if (i < n) dst[i] = (_Float16)src[i];
}

// ---------------------------------------------------------------------------
// Furthest point sampling: one block (1024 threads) per batch. First-index
// tie-break argmax to match jnp.argmax.
// ---------------------------------------------------------------------------
__global__ void fps_kernel(const float* __restrict__ xyz, int* __restrict__ cidx,
                           int Npts, int Msamp) {
  __shared__ float dist[Nc];
  __shared__ float rv[1024];
  __shared__ int   ri[1024];
  __shared__ int   s_last;
  int b = blockIdx.x, t = threadIdx.x, T = blockDim.x;
  const float* xp = xyz + (size_t)b * Npts * 3;
  for (int i = t; i < Npts; i += T) dist[i] = 1e10f;
  if (t == 0) { cidx[b * Msamp] = 0; s_last = 0; }
  __syncthreads();
  for (int it = 1; it < Msamp; ++it) {
    int last = s_last;
    float lx = xp[last * 3], ly = xp[last * 3 + 1], lz = xp[last * 3 + 2];
    float bv = -1.f; int bi = 0;
    for (int i = t; i < Npts; i += T) {
      float dx = xp[i * 3] - lx, dy = xp[i * 3 + 1] - ly, dz = xp[i * 3 + 2] - lz;
      float d = fminf(dist[i], dx * dx + dy * dy + dz * dz);
      dist[i] = d;
      if (d > bv) { bv = d; bi = i; }   // strict > keeps lowest index on ties
    }
    rv[t] = bv; ri[t] = bi;
    __syncthreads();
    for (int s = T >> 1; s > 0; s >>= 1) {
      if (t < s) {
        float v2 = rv[t + s]; int i2 = ri[t + s];
        if (v2 > rv[t] || (v2 == rv[t] && i2 < ri[t])) { rv[t] = v2; ri[t] = i2; }
      }
      __syncthreads();
    }
    if (t == 0) { s_last = ri[0]; cidx[b * Msamp + it] = ri[0]; }
    __syncthreads();
  }
}

// ---------------------------------------------------------------------------
// Gather after FPS: new_xyz (ws + d_out), new_comp (int ws + float d_out)
// ---------------------------------------------------------------------------
__global__ void gather_fps(const float* __restrict__ xyz, const int* __restrict__ comp,
                           const int* __restrict__ cidx, float* __restrict__ nxyz,
                           int* __restrict__ ncomp, float* __restrict__ out) {
  int q = blockIdx.x * blockDim.x + threadIdx.x;
  if (q >= Bc * Mc) return;
  int b = q / Mc, n = cidx[q];
  const float* xp = xyz + ((size_t)b * Nc + n) * 3;
  float x = xp[0], y = xp[1], z = xp[2];
  nxyz[q * 3] = x; nxyz[q * 3 + 1] = y; nxyz[q * 3 + 2] = z;
  out[q * 3] = x; out[q * 3 + 1] = y; out[q * 3 + 2] = z;   // new_xyz output
  int cv = comp[b * Nc + n];
  ncomp[q] = cv;
  out[6144 + 524288 + q] = (float)cv;                        // new_comp output
}

__global__ void gather_center(const float* __restrict__ feat1, const int* __restrict__ cidx,
                              float* __restrict__ center) {
  int q = blockIdx.x, c = threadIdx.x;
  int b = q / Mc;
  center[(size_t)q * 64 + c] = feat1[((size_t)b * Nc + cidx[q]) * 64 + c];
}

// ---------------------------------------------------------------------------
// KNN: one thread per query, insertion-sorted top-K. Optional component
// penalty (+1e9) to match knnquery-with-comp.
// ---------------------------------------------------------------------------
template <int K>
__global__ void knn_kernel(const float* __restrict__ ref, int Nr,
                           const float* __restrict__ qry, int Nq,
                           const int* __restrict__ rcomp, const int* __restrict__ qcomp,
                           int* __restrict__ outIdx) {
  int t = blockIdx.x * blockDim.x + threadIdx.x;
  if (t >= Bc * Nq) return;
  int b = t / Nq;
  const float* qp = qry + (size_t)t * 3;
  float qx = qp[0], qy = qp[1], qz = qp[2];
  int qc = qcomp ? qcomp[t] : 0;
  float bd[K]; int bi[K];
#pragma unroll
  for (int k = 0; k < K; ++k) { bd[k] = 3e38f; bi[k] = 0; }
  const float* rp = ref + (size_t)b * Nr * 3;
  const int* rc = rcomp ? rcomp + (size_t)b * Nr : nullptr;
  for (int n = 0; n < Nr; ++n) {
    float dx = rp[n * 3] - qx, dy = rp[n * 3 + 1] - qy, dz = rp[n * 3 + 2] - qz;
    float d = dx * dx + dy * dy + dz * dz;
    if (rc && rc[n] != qc) d += 1e9f;
    if (d < bd[K - 1]) {
      int j = K - 1;
      while (j > 0 && bd[j - 1] > d) { bd[j] = bd[j - 1]; bi[j] = bi[j - 1]; --j; }
      bd[j] = d; bi[j] = n;
    }
  }
  int* op = outIdx + (size_t)t * K;
  for (int k = 0; k < K; ++k) op[k] = bi[k];
}

// ---------------------------------------------------------------------------
// Stage-1 GEMM input: rows = B*M*K1, cols 96 = [d_xyz(3) | o_fea1-center(64) | pad]
// ---------------------------------------------------------------------------
__global__ void prep_A1(const float* __restrict__ xyz, const float* __restrict__ nxyz,
                        const float* __restrict__ feat1, const float* __restrict__ center,
                        const int* __restrict__ idx1, _Float16* __restrict__ A) {
  int r = blockIdx.x * blockDim.x + threadIdx.x;
  if (r >= Bc * Mc * K1c) return;
  int q = r / K1c, b = q / Mc;
  int n = idx1[r];
  _Float16* dst = A + (size_t)r * 96;
  const float* xp = xyz + ((size_t)b * Nc + n) * 3;
  const float* np_ = nxyz + (size_t)q * 3;
  dst[0] = (_Float16)(xp[0] - np_[0]);
  dst[1] = (_Float16)(xp[1] - np_[1]);
  dst[2] = (_Float16)(xp[2] - np_[2]);
  const float* fo = feat1 + ((size_t)b * Nc + n) * 64;
  const float* fc = center + (size_t)q * 64;
#pragma unroll 8
  for (int c = 0; c < 64; ++c) dst[3 + c] = (_Float16)(fo[c] - fc[c]);
  for (int c = 67; c < 96; ++c) dst[c] = (_Float16)0.f;
}

// Stage-1 reduce: c_fea1 = max_k w1, g_fea1 = sum_k concat(d1m, o_fea1)*softmax_k(w1)
__global__ void stage1_reduce(const float* __restrict__ Gout, const int* __restrict__ idx1,
                              const float* __restrict__ xyz, const float* __restrict__ nxyz,
                              const float* __restrict__ feat1,
                              const float* __restrict__ Wdx1, const float* __restrict__ bdx1,
                              float* __restrict__ c1, float* __restrict__ g1) {
  int q = blockIdx.x, c = threadIdx.x;   // c in [0,96)
  int b = q / Mc;
  const float* grow = Gout + (size_t)q * K1c * 96 + c;
  float vals[K1c]; float mx = -1e30f;
#pragma unroll
  for (int k = 0; k < K1c; ++k) { float v = grow[k * 96]; vals[k] = v; mx = fmaxf(mx, v); }
  float se = 0.f;
#pragma unroll
  for (int k = 0; k < K1c; ++k) { float e = __expf(vals[k] - mx); vals[k] = e; se += e; }
  float inv = 1.f / se;
  float w0 = 0, w1 = 0, w2 = 0, bb = 0, nx = 0, ny = 0, nz = 0;
  if (c < 32) {
    w0 = Wdx1[c * 3]; w1 = Wdx1[c * 3 + 1]; w2 = Wdx1[c * 3 + 2]; bb = bdx1[c];
    nx = nxyz[q * 3]; ny = nxyz[q * 3 + 1]; nz = nxyz[q * 3 + 2];
  }
  const int* ip = idx1 + (size_t)q * K1c;
  float acc = 0.f;
  for (int k = 0; k < K1c; ++k) {
    int n = ip[k];
    float mult;
    if (c < 32) {
      const float* xp = xyz + ((size_t)b * Nc + n) * 3;
      float t = (xp[0] - nx) * w0 + (xp[1] - ny) * w1 + (xp[2] - nz) * w2 + bb;
      mult = t > 0.f ? t : 0.f;
    } else {
      mult = feat1[((size_t)b * Nc + n) * 64 + (c - 32)];
    }
    acc += mult * vals[k];
  }
  c1[(size_t)q * 96 + c] = mx;
  g1[(size_t)q * 96 + c] = acc * inv;
}

// Stage-2 GEMM input: rows = B*M*K2, cols 128 = [o_fea2-g1(96) | d_xyz(3) | pad]
__global__ void prep_A2(const float* __restrict__ nxyz, const float* __restrict__ g1,
                        const int* __restrict__ idx2, _Float16* __restrict__ A) {
  int r = blockIdx.x * blockDim.x + threadIdx.x;
  if (r >= Bc * Mc * K2c) return;
  int q = r / K2c, b = q / Mc;
  int n = idx2[r];
  _Float16* dst = A + (size_t)r * 128;
  const float* go = g1 + ((size_t)(b * Mc + n)) * 96;
  const float* gm = g1 + (size_t)q * 96;
#pragma unroll 8
  for (int c = 0; c < 96; ++c) dst[c] = (_Float16)(go[c] - gm[c]);
  const float* xo = nxyz + ((size_t)(b * Mc + n)) * 3;
  const float* xm = nxyz + (size_t)q * 3;
  dst[96] = (_Float16)(xo[0] - xm[0]);
  dst[97] = (_Float16)(xo[1] - xm[1]);
  dst[98] = (_Float16)(xo[2] - xm[2]);
  for (int c = 99; c < 128; ++c) dst[c] = (_Float16)0.f;
}

__global__ void stage2_reduce(const float* __restrict__ Gout, const int* __restrict__ idx2,
                              const float* __restrict__ nxyz, const float* __restrict__ g1,
                              const float* __restrict__ Wdx2, const float* __restrict__ bdx2,
                              float* __restrict__ c2, float* __restrict__ g2) {
  int q = blockIdx.x, c = threadIdx.x;   // c in [0,128)
  int b = q / Mc;
  const float* grow = Gout + (size_t)q * K2c * 128 + c;
  float vals[K2c]; float mx = -1e30f;
#pragma unroll
  for (int k = 0; k < K2c; ++k) { float v = grow[k * 128]; vals[k] = v; mx = fmaxf(mx, v); }
  float se = 0.f;
#pragma unroll
  for (int k = 0; k < K2c; ++k) { float e = __expf(vals[k] - mx); vals[k] = e; se += e; }
  float inv = 1.f / se;
  float w0 = 0, w1 = 0, w2 = 0, bb = 0, nx = 0, ny = 0, nz = 0;
  if (c < 32) {
    w0 = Wdx2[c * 3]; w1 = Wdx2[c * 3 + 1]; w2 = Wdx2[c * 3 + 2]; bb = bdx2[c];
    nx = nxyz[q * 3]; ny = nxyz[q * 3 + 1]; nz = nxyz[q * 3 + 2];
  }
  const int* ip = idx2 + (size_t)q * K2c;
  float acc = 0.f;
  for (int k = 0; k < K2c; ++k) {
    int n = ip[k];
    float mult;
    if (c < 32) {
      const float* xp = nxyz + ((size_t)(b * Mc + n)) * 3;
      float t = (xp[0] - nx) * w0 + (xp[1] - ny) * w1 + (xp[2] - nz) * w2 + bb;
      mult = t > 0.f ? t : 0.f;
    } else {
      mult = g1[((size_t)(b * Mc + n)) * 96 + (c - 32)];
    }
    acc += mult * vals[k];
  }
  c2[(size_t)q * 128 + c] = mx;
  g2[(size_t)q * 128 + c] = acc * inv;
}

// Stage-3b GEMM input: rows = B*M*K2, cols 160 = [o_fea3-g2(128) | d_xyz(3) | pad]
__global__ void prep_A3b(const float* __restrict__ nxyz, const float* __restrict__ g2,
                         const int* __restrict__ idx2, _Float16* __restrict__ A) {
  int r = blockIdx.x * blockDim.x + threadIdx.x;
  if (r >= Bc * Mc * K2c) return;
  int q = r / K2c, b = q / Mc;
  int n = idx2[r];
  _Float16* dst = A + (size_t)r * 160;
  const float* go = g2 + ((size_t)(b * Mc + n)) * 128;
  const float* gm = g2 + (size_t)q * 128;
#pragma unroll 8
  for (int c = 0; c < 128; ++c) dst[c] = (_Float16)(go[c] - gm[c]);
  const float* xo = nxyz + ((size_t)(b * Mc + n)) * 3;
  const float* xm = nxyz + (size_t)q * 3;
  dst[128] = (_Float16)(xo[0] - xm[0]);
  dst[129] = (_Float16)(xo[1] - xm[1]);
  dst[130] = (_Float16)(xo[2] - xm[2]);
  for (int c = 131; c < 160; ++c) dst[c] = (_Float16)0.f;
}

__global__ void max_over_k(const float* __restrict__ Gout, float* __restrict__ lpf) {
  int q = blockIdx.x, c = threadIdx.x;   // c in [0,128)
  float mx = -1e30f;
#pragma unroll
  for (int k = 0; k < K2c; ++k) mx = fmaxf(mx, Gout[((size_t)q * K2c + k) * 128 + c]);
  lpf[(size_t)q * 128 + c] = mx;
}

// fea3 = [g_fea3(=c2) | lpf | g2 | c2 | g1 | c1 | center] -> fp16 (2048 x 768)
__global__ void prep_fea3(const float* __restrict__ c2, const float* __restrict__ lpf,
                          const float* __restrict__ g2, const float* __restrict__ g1,
                          const float* __restrict__ c1, const float* __restrict__ center,
                          _Float16* __restrict__ A) {
  int q = blockIdx.x, c = threadIdx.x;   // c in [0,768)
  float v;
  if      (c < 128) v = c2[(size_t)q * 128 + c];            // g_fea3 == c_fea2 (softmax sums to 1)
  else if (c < 256) v = lpf[(size_t)q * 128 + (c - 128)];
  else if (c < 384) v = g2[(size_t)q * 128 + (c - 256)];
  else if (c < 512) v = c2[(size_t)q * 128 + (c - 384)];
  else if (c < 608) v = g1[(size_t)q * 96 + (c - 512)];
  else if (c < 704) v = c1[(size_t)q * 96 + (c - 608)];
  else              v = center[(size_t)q * 64 + (c - 704)];
  A[(size_t)q * 768 + c] = (_Float16)v;
}

// new_features transpose: (B, M, 256) -> (B, 256, M) into d_out
__global__ void write_feats(const float* __restrict__ nf, float* __restrict__ out) {
  int i = blockIdx.x * blockDim.x + threadIdx.x;
  if (i >= Bc * 256 * Mc) return;
  int b = i / (256 * Mc), rem = i % (256 * Mc);
  int c = rem / Mc, m = rem % Mc;
  out[6144 + i] = nf[((size_t)b * Mc + m) * 256 + c];
}

// ---------------------------------------------------------------------------
extern "C" void kernel_launch(void* const* d_in, const int* in_sizes, int n_in,
                              void* d_out, int out_size, void* d_ws, size_t ws_size,
                              hipStream_t stream) {
  (void)in_sizes; (void)n_in; (void)out_size; (void)ws_size;
  const float* xyz   = (const float*)d_in[0];
  const float* feats = (const float*)d_in[1];
  const int*   comp  = (const int*)d_in[2];
  const float* W1d  = (const float*)d_in[3];  const float* b1d  = (const float*)d_in[4];
  const float* Wdx1 = (const float*)d_in[5];  const float* bdx1 = (const float*)d_in[6];
  const float* Ww1  = (const float*)d_in[7];  const float* bw1  = (const float*)d_in[8];
  const float* Wdx2 = (const float*)d_in[9];  const float* bdx2 = (const float*)d_in[10];
  const float* Ww2  = (const float*)d_in[11]; const float* bw2  = (const float*)d_in[12];
  // d_in[13]/d_in[14] = Ww3/bw3: provably unused (g_fea3 == c_fea2).
  const float* Wsp3 = (const float*)d_in[15]; const float* bsp3 = (const float*)d_in[16];
  const float* Wnew = (const float*)d_in[17]; const float* bnew = (const float*)d_in[18];
  float* out = (float*)d_out;

  // Workspace bump allocator (256B aligned)
  char* ws = (char*)d_ws;
  size_t off = 0;
  auto alloc = [&](size_t bytes) -> void* {
    off = (off + 255) & ~(size_t)255;
    void* p = ws + off; off += bytes; return p;
  };
  int*   cidx   = (int*)alloc((size_t)Bc * Mc * 4);
  int*   ncomp  = (int*)alloc((size_t)Bc * Mc * 4);
  int*   idx1   = (int*)alloc((size_t)Bc * Mc * K1c * 4);
  int*   idx2   = (int*)alloc((size_t)Bc * Mc * K2c * 4);
  float* nxyz   = (float*)alloc((size_t)Bc * Mc * 3 * 4);
  float* feat1  = (float*)alloc((size_t)Bc * Nc * 64 * 4);
  float* center = (float*)alloc((size_t)Bc * Mc * 64 * 4);
  float* g1     = (float*)alloc((size_t)Bc * Mc * 96 * 4);
  float* c1     = (float*)alloc((size_t)Bc * Mc * 96 * 4);
  float* g2     = (float*)alloc((size_t)Bc * Mc * 128 * 4);
  float* c2     = (float*)alloc((size_t)Bc * Mc * 128 * 4);
  float* lpf    = (float*)alloc((size_t)Bc * Mc * 128 * 4);
  float* newf   = (float*)alloc((size_t)Bc * Mc * 256 * 4);
  _Float16* w1dh  = (_Float16*)alloc((size_t)64 * 64 * 2);
  _Float16* ww1h  = (_Float16*)alloc((size_t)96 * 96 * 2);
  _Float16* ww2h  = (_Float16*)alloc((size_t)128 * 128 * 2);
  _Float16* wsp3h = (_Float16*)alloc((size_t)128 * 160 * 2);
  _Float16* wnewh = (_Float16*)alloc((size_t)256 * 768 * 2);
  _Float16* Ah    = (_Float16*)alloc((size_t)Bc * Mc * K1c * 96 * 2);  // largest A (reused)
  float*    Gout  = (float*)alloc((size_t)Bc * Mc * K1c * 96 * 4);     // largest GEMM out (reused)

  auto cdiv = [](int a, int b) { return (a + b - 1) / b; };

  // 1. Pack weights to padded fp16
  pack_w<<<cdiv(64 * 64, 256), 256, 0, stream>>>(W1d,  w1dh,  64, 64, 64);
  pack_w<<<cdiv(96 * 96, 256), 256, 0, stream>>>(Ww1,  ww1h,  96, 67, 96);
  pack_w<<<cdiv(128 * 128, 256), 256, 0, stream>>>(Ww2,  ww2h,  128, 99, 128);
  pack_w<<<cdiv(128 * 160, 256), 256, 0, stream>>>(Wsp3, wsp3h, 128, 131, 160);
  pack_w<<<cdiv(256 * 768, 256), 256, 0, stream>>>(Wnew, wnewh, 256, 768, 768);

  // 2. feat1 = ReLU(features @ W1d^T + b1d)  [WMMA]
  f32_to_f16<<<cdiv(Bc * Nc * 64, 256), 256, 0, stream>>>(feats, Ah, Bc * Nc * 64);
  {
    int rows = Bc * Nc, jobs = (rows / 16) * (64 / 16);
    gemm_relu_wmma<<<cdiv(jobs, 8), 256, 0, stream>>>(Ah, w1dh, b1d, feat1, rows, 64, 64);
  }

  // 3. FPS + gathers (also writes new_xyz / new_comp outputs)
  fps_kernel<<<Bc, 1024, 0, stream>>>(xyz, cidx, Nc, Mc);
  gather_fps<<<cdiv(Bc * Mc, 256), 256, 0, stream>>>(xyz, comp, cidx, nxyz, ncomp, out);
  gather_center<<<Bc * Mc, 64, 0, stream>>>(feat1, cidx, center);

  // 4. KNNs (idx3 == idx2: identical deterministic _knn call, reused)
  knn_kernel<K1c><<<cdiv(Bc * Mc, 128), 128, 0, stream>>>(xyz, Nc, nxyz, Mc, nullptr, nullptr, idx1);
  knn_kernel<K2c><<<cdiv(Bc * Mc, 128), 128, 0, stream>>>(nxyz, Mc, nxyz, Mc, ncomp, ncomp, idx2);

  // 5. Stage 1: w1 GEMM [WMMA] + softmax-weighted reduce
  prep_A1<<<cdiv(Bc * Mc * K1c, 256), 256, 0, stream>>>(xyz, nxyz, feat1, center, idx1, Ah);
  {
    int rows = Bc * Mc * K1c, jobs = (rows / 16) * (96 / 16);
    gemm_relu_wmma<<<cdiv(jobs, 8), 256, 0, stream>>>(Ah, ww1h, bw1, Gout, rows, 96, 96);
  }
  stage1_reduce<<<Bc * Mc, 96, 0, stream>>>(Gout, idx1, xyz, nxyz, feat1, Wdx1, bdx1, c1, g1);

  // 6. Stage 2: w2 GEMM [WMMA] + reduce
  prep_A2<<<cdiv(Bc * Mc * K2c, 256), 256, 0, stream>>>(nxyz, g1, idx2, Ah);
  {
    int rows = Bc * Mc * K2c, jobs = (rows / 16) * (128 / 16);
    gemm_relu_wmma<<<cdiv(jobs, 8), 256, 0, stream>>>(Ah, ww2h, bw2, Gout, rows, 128, 128);
  }
  stage2_reduce<<<Bc * Mc, 128, 0, stream>>>(Gout, idx2, nxyz, g1, Wdx2, bdx2, c2, g2);

  // 7. Stage 3b: Wsp3 GEMM [WMMA] + max over K
  prep_A3b<<<cdiv(Bc * Mc * K2c, 256), 256, 0, stream>>>(nxyz, g2, idx2, Ah);
  {
    int rows = Bc * Mc * K2c, jobs = (rows / 16) * (128 / 16);
    gemm_relu_wmma<<<cdiv(jobs, 8), 256, 0, stream>>>(Ah, wsp3h, bsp3, Gout, rows, 160, 128);
  }
  max_over_k<<<Bc * Mc, 128, 0, stream>>>(Gout, lpf);

  // 8. Final projection: 768 -> 256 [WMMA], transpose to (B,256,M)
  prep_fea3<<<Bc * Mc, 768, 0, stream>>>(c2, lpf, g2, g1, c1, center, Ah);
  {
    int rows = Bc * Mc, jobs = (rows / 16) * (256 / 16);
    gemm_relu_wmma<<<cdiv(jobs, 8), 256, 0, stream>>>(Ah, wnewh, bnew, newf, rows, 768, 256);
  }
  write_feats<<<cdiv(Bc * 256 * Mc, 256), 256, 0, stream>>>(newf, out);
}
